// SparseAttention_72507637891549
// MI455X (gfx1250) — compile-verified, matrix operand order checked by reading, not run on hardware
//
#include <hip/hip_runtime.h>
#include <hip/hip_bf16.h>

typedef __attribute__((ext_vector_type(16))) _Float16 v16h;
typedef __attribute__((ext_vector_type(8)))  _Float16 v8h;
typedef __attribute__((ext_vector_type(4)))  _Float16 v4h;
typedef __attribute__((ext_vector_type(8)))  float    v8f;
typedef __attribute__((ext_vector_type(4))) unsigned int u32x4;
typedef __attribute__((ext_vector_type(8))) int          i32x8;
typedef __attribute__((ext_vector_type(4))) int          i32x4;

#define D_MODEL   1024
#define N_HEADS   16
#define HEAD_DIM  64
#define HALF_W    32
#define CHUNKQ    64
#define BATCH     4
#define SEQ       4096
#define MTOT      (BATCH * SEQ)   // 16384

#if __has_builtin(__builtin_amdgcn_tensor_load_to_lds)
#define HAVE_TDM 1
#else
#define HAVE_TDM 0
#endif

// ---------------------------------------------------------------------------
// WMMA 16-bit fragment loader (A and B share the same per-lane layout):
//   lanes 0-15  : row/col = lane,    K = {0..7, 16..23}
//   lanes 16-31 : row/col = lane-16, K = {8..15, 24..31}
// => two contiguous 16B LDS reads per lane.
// ---------------------------------------------------------------------------
__device__ __forceinline__ v16h ldfrag(const _Float16* base, int ld, int lane) {
    const int lr = lane & 15;
    const int hf = lane >> 4;
    const _Float16* p = base + lr * ld + hf * 8;
    union { v16h v; v8h h[2]; } u;
    u.h[0] = *(const v8h*)(p);        // K = hf*8 .. hf*8+7
    u.h[1] = *(const v8h*)(p + 16);   // K = hf*8+16 .. hf*8+23
    return u.v;
}

#if HAVE_TDM
// LDS byte offset of a __shared__ object: generic shared address = aperture
// base in [63:32], LDS offset in [31:0] (ISA 10.2 aperture mapping).
__device__ __forceinline__ unsigned lds_off(const void* p) {
    return (unsigned)(size_t)p;
}

// TDM descriptor per CDNA5 ISA:
// 2-D f16 tile: tile_dim0=32 (contiguous K), tile_dim1=rows, row stride=K.
// LDS padding: +4 DWORDs (8 f16) after every 16 DWORDs (32 f16) -> ld 40.
__device__ __forceinline__ void tdm_load_tile_f16(const _Float16* gptr,
                                                  unsigned lds_byte_off,
                                                  unsigned tensor_d0,
                                                  unsigned tensor_d1,
                                                  unsigned tile_d1,
                                                  unsigned stride0) {
    unsigned long long ga = (unsigned long long)(size_t)(const void*)gptr;
    u32x4 g0;
    g0[0] = 1u;                                            // count=1, user D#
    g0[1] = lds_byte_off;                                  // lds_addr
    g0[2] = (unsigned)(ga & 0xFFFFFFFFu);                  // global_addr[31:0]
    g0[3] = (unsigned)((ga >> 32) & 0x1FFFFFFu) | (2u << 30);  // [56:32] | type=2
    i32x8 g1;
    g1[0] = (1 << 16)      // data_size = 2 bytes
          | (1 << 20)      // pad_enable
          | (3 << 22)      // pad_interval: 16 DWORDs
          | (3 << 25);     // pad_amount: 4 DWORDs
    g1[1] = (int)((tensor_d0 & 0xFFFFu) << 16);                        // dim0 lo16
    g1[2] = (int)((tensor_d0 >> 16) | ((tensor_d1 & 0xFFFFu) << 16));  // dim0 hi | dim1 lo
    g1[3] = (int)((tensor_d1 >> 16) | (32u << 16));                    // dim1 hi | tile_dim0=32
    g1[4] = (int)(tile_d1 & 0xFFFFu);                                  // tile_dim1, tile_dim2=0
    g1[5] = (int)stride0;                                              // dim0_stride lo32
    g1[6] = 0;
    g1[7] = 0;
    i32x4 z4 = {0, 0, 0, 0};
    i32x8 z8 = {0, 0, 0, 0, 0, 0, 0, 0};
    // clang-23 / therock-10.0 declares the 6-arg form:
    // (uint32x4, int32x8, int32x4, int32x4, int32x8, i32 cpol)
    __builtin_amdgcn_tensor_load_to_lds(g0, g1, z4, z4, z8, 0);
}

__device__ __forceinline__ void tdm_wait0() {
#if __has_builtin(__builtin_amdgcn_s_wait_tensorcnt)
    __builtin_amdgcn_s_wait_tensorcnt(0);
#else
    asm volatile("s_wait_tensorcnt 0x0" ::: "memory");
#endif
}
#endif // HAVE_TDM

// ---------------------------------------------------------------------------
// fp32 -> f16 conversion (counts % 4 == 0)
// ---------------------------------------------------------------------------
__global__ void swa_cvt_f16(const float* __restrict__ in, _Float16* __restrict__ out, int n) {
    int i = (blockIdx.x * blockDim.x + threadIdx.x) * 4;
    if (i + 3 < n) {
        float4 f = *(const float4*)(in + i);
        v4h o = { (_Float16)f.x, (_Float16)f.y, (_Float16)f.z, (_Float16)f.w };
        *(v4h*)(out + i) = o;
    }
}

// ---------------------------------------------------------------------------
// One K-step of the block tile: wave computes 64x32 (4x2 WMMA tiles).
// ---------------------------------------------------------------------------
__device__ __forceinline__ void mma_step(const _Float16* tA, const _Float16* tB,
                                         v8f acc[4][2], int wm, int wn, int lane) {
    v16h af[4], bf[2];
    #pragma unroll
    for (int i = 0; i < 4; ++i) af[i] = ldfrag(tA + (wm + i * 16) * 40, 40, lane);
    #pragma unroll
    for (int j = 0; j < 2; ++j) bf[j] = ldfrag(tB + (wn + j * 16) * 40, 40, lane);
    #pragma unroll
    for (int i = 0; i < 4; ++i)
        #pragma unroll
        for (int j = 0; j < 2; ++j)
            acc[i][j] = __builtin_amdgcn_wmma_f32_16x16x32_f16(
                false, af[i], false, bf[j], (short)0, acc[i][j], false, false);
}

// ---------------------------------------------------------------------------
// GEMM: C[m,n] = sum_k A[m,k] * W[n,k] + bias[n]
// A: [M,K] f16 row-major, W: [N,K] f16 row-major (both K-contiguous -> direct
// WMMA fragment layouts). 256 threads = 8 waves, 128x128 tile, K-step 32.
// LDS tiles double-buffered and filled by the Tensor Data Mover (wave 0
// issues tensor_load_to_lds for tile t+1 while all waves compute tile t).
// ---------------------------------------------------------------------------
template<bool OUT_F16>
__launch_bounds__(256)
__global__ void swa_gemm_xwT(const _Float16* __restrict__ A,
                             const _Float16* __restrict__ W,
                             const float*    __restrict__ bias,
                             _Float16* __restrict__ outh,
                             float*    __restrict__ outf,
                             int M, int N, int K) {
    __shared__ _Float16 sA[2][128 * 40];   // pad 32 -> 40 f16/row (bank spread)
    __shared__ _Float16 sB[2][128 * 40];

    const int tid  = threadIdx.x;
    const int lane = tid & 31;
    const int wave = tid >> 5;
    const int lr   = lane & 15;
    const int hf   = lane >> 4;
    const int m0   = blockIdx.y * 128;
    const int n0   = blockIdx.x * 128;
    const int wm   = (wave >> 2) * 64;   // wave tile origin in M (0 / 64)
    const int wn   = (wave & 3) * 32;    // wave tile origin in N (0/32/64/96)

    v8f acc[4][2] = {};
    const int nkt = K / 32;

#if HAVE_TDM
    const _Float16* Abase = A + (size_t)m0 * K;
    const _Float16* Bbase = W + (size_t)n0 * K;
    const unsigned ldsA0 = lds_off(&sA[0][0]), ldsA1 = lds_off(&sA[1][0]);
    const unsigned ldsB0 = lds_off(&sB[0][0]), ldsB1 = lds_off(&sB[1][0]);

    if (wave == 0) {   // TDM issue is wave-level (EXEC ignored)
        tdm_load_tile_f16(Abase, ldsA0, (unsigned)K, 128u, 128u, (unsigned)K);
        tdm_load_tile_f16(Bbase, ldsB0, (unsigned)K, 128u, 128u, (unsigned)K);
        tdm_wait0();
    }
    __syncthreads();

    int cur = 0;
    for (int t = 0; t < nkt; ++t) {
        if (wave == 0 && t + 1 < nkt) {   // async prefetch of next K tile
            tdm_load_tile_f16(Abase + (size_t)(t + 1) * 32, cur ? ldsA0 : ldsA1,
                              (unsigned)K, 128u, 128u, (unsigned)K);
            tdm_load_tile_f16(Bbase + (size_t)(t + 1) * 32, cur ? ldsB0 : ldsB1,
                              (unsigned)K, 128u, 128u, (unsigned)K);
        }
        mma_step(&sA[cur][0], &sB[cur][0], acc, wm, wn, lane);
        if (wave == 0 && t + 1 < nkt) tdm_wait0();
        __syncthreads();
        cur ^= 1;
    }
#else
    for (int t = 0; t < nkt; ++t) {
        #pragma unroll
        for (int c = 0; c < 2; ++c) {
            int chunk = c * 256 + tid;          // 0..511
            int row   = chunk >> 2;             // 0..127
            int c8    = (chunk & 3) * 8;        // 0,8,16,24
            *(v8h*)(&sA[0][row * 40 + c8]) =
                *(const v8h*)(A + (size_t)(m0 + row) * K + t * 32 + c8);
            *(v8h*)(&sB[0][row * 40 + c8]) =
                *(const v8h*)(W + (size_t)(n0 + row) * K + t * 32 + c8);
        }
        __syncthreads();
        mma_step(&sA[0][0], &sB[0][0], acc, wm, wn, lane);
        __syncthreads();
    }
#endif

    // epilogue: C/D layout -> element r maps to M = r + 8*hf, N = lr
    #pragma unroll
    for (int i = 0; i < 4; ++i)
        #pragma unroll
        for (int j = 0; j < 2; ++j) {
            int nn = n0 + wn + j * 16 + lr;
            float bv = bias[nn];
            #pragma unroll
            for (int r = 0; r < 8; ++r) {
                int mm  = m0 + wm + i * 16 + hf * 8 + r;
                float v = acc[i][j][r] + bv;
                if (OUT_F16) outh[(size_t)mm * N + nn] = (_Float16)v;
                else         outf[(size_t)mm * N + nn] = v;
            }
        }
}

// ---------------------------------------------------------------------------
// Banded attention: one block per (batch, head, 64-query chunk).
// Key window: 128 keys starting at n*64 - 32 (covers the +/-32 band).
// 4 waves; wave w owns query rows [16w, 16w+16).
// ---------------------------------------------------------------------------
__launch_bounds__(128)
__global__ void swa_attn_band(const _Float16* __restrict__ qh,
                              const _Float16* __restrict__ kh,
                              const _Float16* __restrict__ vh,
                              _Float16* __restrict__ ctxh) {
    __shared__ _Float16 sQ [64 * 72];    // [query][d]   (A-frag for Q*K^T)
    __shared__ _Float16 sK [128 * 72];   // [key][d]     (B-frag for Q*K^T)
    __shared__ _Float16 sVt[64 * 136];   // [d][key]     (B-frag for P*V)
    __shared__ _Float16 sP [64 * 136];   // [query][key] (A-frag for P*V)
    __shared__ float    sS [64 * 128];   // scores

    const int tid  = threadIdx.x;
    const int lane = tid & 31;
    const int wave = tid >> 5;
    const int lr   = lane & 15;
    const int hf   = lane >> 4;
    const int nblk = blockIdx.x;         // chunk index 0..63
    const int h    = blockIdx.y;
    const int b    = blockIdx.z;
    const size_t rowbase  = (size_t)b * SEQ;
    const int    col0     = h * HEAD_DIM;
    const int    keystart = nblk * CHUNKQ - HALF_W;   // may be negative

    // ---- load Q tile: 64 x 64 ----
    #pragma unroll
    for (int c = 0; c < 4; ++c) {
        int chunk = c * 128 + tid;       // 0..511
        int qi = chunk >> 3;             // 0..63
        int c8 = (chunk & 7) * 8;        // 0..56
        *(v8h*)(sQ + qi * 72 + c8) =
            *(const v8h*)(qh + (rowbase + nblk * CHUNKQ + qi) * D_MODEL + col0 + c8);
    }
    // ---- load K (row-major) and V (transposed) tiles: 128 x 64 ----
    #pragma unroll
    for (int c = 0; c < 8; ++c) {
        int chunk = c * 128 + tid;       // 0..1023
        int kj = chunk >> 3;             // 0..127
        int c8 = (chunk & 7) * 8;
        int keypos = keystart + kj;
        v8h kv = {};
        v8h vv = {};
        if (keypos >= 0 && keypos < SEQ) {
            kv = *(const v8h*)(kh + (rowbase + keypos) * D_MODEL + col0 + c8);
            vv = *(const v8h*)(vh + (rowbase + keypos) * D_MODEL + col0 + c8);
        }
        *(v8h*)(sK + kj * 72 + c8) = kv;
        #pragma unroll
        for (int t = 0; t < 8; ++t) sVt[(c8 + t) * 136 + kj] = vv[t];
    }
    __syncthreads();

    // ---- S = Q*K^T (strip of 16 queries x 128 keys per wave) ----
    const int q0 = wave * 16;
    const float scale = 0.125f;          // 1/sqrt(64)
    #pragma unroll
    for (int tn = 0; tn < 8; ++tn) {
        v8f acc = {};
        #pragma unroll
        for (int kk = 0; kk < HEAD_DIM; kk += 32) {
            v16h a  = ldfrag(sQ + q0 * 72 + kk, 72, lane);
            v16h bf = ldfrag(sK + tn * 16 * 72 + kk, 72, lane);
            acc = __builtin_amdgcn_wmma_f32_16x16x32_f16(
                false, a, false, bf, (short)0, acc, false, false);
        }
        int kcol   = tn * 16 + lr;
        int keypos = keystart + kcol;
        #pragma unroll
        for (int r = 0; r < 8; ++r) {
            int qi = q0 + hf * 8 + r;
            // band: -32 <= keypos - qpos <= 32  <=>  qi <= kcol <= qi + 64
            bool valid = (kcol >= qi) && (kcol <= qi + 2 * HALF_W) &&
                         (keypos >= 0) && (keypos < SEQ);
            sS[qi * 128 + kcol] = valid ? acc[r] * scale : -3.0e38f;
        }
    }
    __syncthreads();

    // ---- softmax: one row per thread ----
    if (tid < 64) {
        float mx = -3.0e38f;
        for (int j = 0; j < 128; ++j) mx = fmaxf(mx, sS[tid * 128 + j]);
        float sum = 0.f;
        for (int j = 0; j < 128; ++j) {
            float e = __expf(sS[tid * 128 + j] - mx);
            sS[tid * 128 + j] = e;
            sum += e;
        }
        float inv = 1.f / sum;
        for (int j = 0; j < 128; ++j)
            sP[tid * 136 + j] = (_Float16)(sS[tid * 128 + j] * inv);
    }
    __syncthreads();

    // ---- ctx = P * V (strip of 16 queries x 64 dims per wave) ----
    #pragma unroll
    for (int tn = 0; tn < 4; ++tn) {
        v8f acc = {};
        #pragma unroll
        for (int kk = 0; kk < 128; kk += 32) {
            v16h a  = ldfrag(sP + q0 * 136 + kk, 136, lane);
            v16h bf = ldfrag(sVt + tn * 16 * 136 + kk, 136, lane);
            acc = __builtin_amdgcn_wmma_f32_16x16x32_f16(
                false, a, false, bf, (short)0, acc, false, false);
        }
        #pragma unroll
        for (int r = 0; r < 8; ++r) {
            int qi = q0 + hf * 8 + r;
            int d  = tn * 16 + lr;
            ctxh[(rowbase + nblk * CHUNKQ + qi) * D_MODEL + col0 + d] = (_Float16)acc[r];
        }
    }
}

// ---------------------------------------------------------------------------
extern "C" void kernel_launch(void* const* d_in, const int* in_sizes, int n_in,
                              void* d_out, int out_size, void* d_ws, size_t ws_size,
                              hipStream_t stream) {
    const float* x  = (const float*)d_in[0];
    const float* Wq = (const float*)d_in[1];
    const float* bq = (const float*)d_in[2];
    const float* Wk = (const float*)d_in[3];
    const float* bk = (const float*)d_in[4];
    const float* Wv = (const float*)d_in[5];
    const float* bv = (const float*)d_in[6];
    const float* Wo = (const float*)d_in[7];
    const float* bo = (const float*)d_in[8];
    float* out = (float*)d_out;

    const size_t X_ELEMS = (size_t)MTOT * D_MODEL;       // 16M
    const size_t W_ELEMS = (size_t)D_MODEL * D_MODEL;    // 1M

    char* ws = (char*)d_ws;
    _Float16* xh   = (_Float16*)ws; ws += X_ELEMS * sizeof(_Float16);
    _Float16* Wqh  = (_Float16*)ws; ws += W_ELEMS * sizeof(_Float16);
    _Float16* Wkh  = (_Float16*)ws; ws += W_ELEMS * sizeof(_Float16);
    _Float16* Wvh  = (_Float16*)ws; ws += W_ELEMS * sizeof(_Float16);
    _Float16* Woh  = (_Float16*)ws; ws += W_ELEMS * sizeof(_Float16);
    _Float16* qh   = (_Float16*)ws; ws += X_ELEMS * sizeof(_Float16);
    _Float16* kh   = (_Float16*)ws; ws += X_ELEMS * sizeof(_Float16);
    _Float16* vh   = (_Float16*)ws; ws += X_ELEMS * sizeof(_Float16);
    _Float16* ctxh = (_Float16*)ws; ws += X_ELEMS * sizeof(_Float16);

    // 1) fp32 -> f16 (x_h = 32 MB: resident in 192 MB L2 for all 3 QKV GEMMs)
    swa_cvt_f16<<<dim3((unsigned)(X_ELEMS / 4 / 256)), 256, 0, stream>>>(x,  xh,  (int)X_ELEMS);
    swa_cvt_f16<<<dim3((unsigned)(W_ELEMS / 4 / 256)), 256, 0, stream>>>(Wq, Wqh, (int)W_ELEMS);
    swa_cvt_f16<<<dim3((unsigned)(W_ELEMS / 4 / 256)), 256, 0, stream>>>(Wk, Wkh, (int)W_ELEMS);
    swa_cvt_f16<<<dim3((unsigned)(W_ELEMS / 4 / 256)), 256, 0, stream>>>(Wv, Wvh, (int)W_ELEMS);
    swa_cvt_f16<<<dim3((unsigned)(W_ELEMS / 4 / 256)), 256, 0, stream>>>(Wo, Woh, (int)W_ELEMS);

    // 2) Q/K/V projections (f16 WMMA, f32 accumulate, TDM-fed 128x128 tiles)
    dim3 gg(D_MODEL / 128, MTOT / 128);   // (8, 128)
    swa_gemm_xwT<true><<<gg, 256, 0, stream>>>(xh, Wqh, bq, qh, nullptr, MTOT, D_MODEL, D_MODEL);
    swa_gemm_xwT<true><<<gg, 256, 0, stream>>>(xh, Wkh, bk, kh, nullptr, MTOT, D_MODEL, D_MODEL);
    swa_gemm_xwT<true><<<gg, 256, 0, stream>>>(xh, Wvh, bv, vh, nullptr, MTOT, D_MODEL, D_MODEL);

    // 3) banded attention
    swa_attn_band<<<dim3(SEQ / CHUNKQ, N_HEADS, BATCH), 128, 0, stream>>>(qh, kh, vh, ctxh);

    // 4) output projection (f32 out)
    swa_gemm_xwT<false><<<gg, 256, 0, stream>>>(ctxh, Woh, bo, nullptr, out, MTOT, D_MODEL, D_MODEL);
}
